// GraphHead_1322849927528
// MI455X (gfx1250) — compile-verified, matrix operand order checked by reading, not run on hardware
//
#include <hip/hip_runtime.h>
#include <hip/hip_bf16.h>
#include <math.h>

// ---------------------------------------------------------------------------
// MI455X / gfx1250 implementation.
// All dense math runs through v_wmma_f32_16x16x32_bf16 (f32 in -> bf16 tiles
// in LDS -> f32 accumulators). wave32; 256-thread blocks (8 waves); each block
// computes a 128x64 output tile, each wave a 32x32 sub-tile (4 accumulators,
// 8 WMMAs per LDS stage of K=64).
//
// The 16-expert contractions are fused into single K=1024 GEMMs:
//   sum_k m_k @ W3_k == m_all @ W3stacked   (W3 (16,64,1024) is already the
//   stacked (1024,1024) row-major matrix), with s_all = sp @ W2' where W2' is
//   a one-time (k,a)->(a,k) repack. Eliminates all GF_ACC output sweeps.
//
// Workspace requirement: ~340 MB of f32 scratch.
// ---------------------------------------------------------------------------

typedef __attribute__((ext_vector_type(16))) __bf16 bf16x16;
typedef __attribute__((ext_vector_type(8)))  __bf16 bf16x8;
typedef __attribute__((ext_vector_type(8)))  float  floatx8;

#define GF_RELU 1
#define GF_ACC  2
#define GF_BIAS 4

__device__ __forceinline__ unsigned short f2bf_bits(float f) {
  union { float f; unsigned u; } a; a.f = f;
  unsigned r = a.u + 0x7FFFu + ((a.u >> 16) & 1u);   // round-to-nearest-even
  return (unsigned short)(r >> 16);
}

__device__ __forceinline__ bf16x16 frag_from_lds(const unsigned short* p) {
  bf16x8 lo = *(const bf16x8*)p;
  bf16x8 hi = *(const bf16x8*)(p + 16);
  return __builtin_shufflevector(lo, hi, 0,1,2,3,4,5,6,7,8,9,10,11,12,13,14,15);
}

// ---------------------------------------------------------------------------
// Generic GEMM: C[M,N] = op( A[M,K](f32,lda) * B[K,N](f32,ldb) [+bias] [+C] )
// ---------------------------------------------------------------------------
__global__ __launch_bounds__(256)
void gemm_wmma_kernel(const float* __restrict__ A, const float* __restrict__ B,
                      float* __restrict__ C, const float* __restrict__ bias,
                      int M, int N, int K, int lda, int ldb, int ldc, int flags)
{
  __shared__ unsigned short As[128][64];   // [m][k] bf16 bits  (16 KB)
  __shared__ unsigned short Bs[64][64];    // [n][k] bf16 bits  ( 8 KB)

  const int tid  = threadIdx.x;
  const int lane = tid & 31;
  const int wid  = tid >> 5;              // 0..7
  const int m0   = blockIdx.y * 128;
  const int n0   = blockIdx.x * 64;
  const int wm   = (wid >> 1) * 32;       // 0,32,64,96
  const int wn   = (wid & 1) * 32;        // 0,32

  floatx8 acc00 = {}, acc01 = {}, acc10 = {}, acc11 = {};

  const int mrow = lane & 15;             // fragment row/col within 16
  const int kb   = (lane >> 4) * 8;       // 0 or 8

  const bool fullM = (m0 + 128 <= M);
  const bool fullN = (n0 + 64 <= N);

  for (int k0 = 0; k0 < K; k0 += 64) {
    if (fullM && fullN && (k0 + 64 <= K)) {
      // ---- fast path: guard-free vector loads ----
      {
        int row = tid >> 1;
        int ks  = (tid & 1) * 32;
        const float* ap = A + (size_t)(m0 + row) * lda + k0 + ks;
        #pragma unroll
        for (int j = 0; j < 16; ++j) {
          float2 v = *(const float2*)(ap + j * 2);
          unsigned pk = (unsigned)f2bf_bits(v.x) | ((unsigned)f2bf_bits(v.y) << 16);
          *(unsigned*)&As[row][ks + j * 2] = pk;
        }
      }
      {
        int kp = (tid >> 3) * 2;          // k pair base (0,2,..,62)
        int nc = (tid & 7) * 8;           // 8 consecutive n
        const float* b0 = B + (size_t)(k0 + kp) * ldb + n0 + nc;
        const float* b1 = b0 + ldb;
        float2 r0[4], r1[4];
        #pragma unroll
        for (int j = 0; j < 4; ++j) {
          r0[j] = *(const float2*)(b0 + j * 2);
          r1[j] = *(const float2*)(b1 + j * 2);
        }
        #pragma unroll
        for (int j = 0; j < 4; ++j) {
          unsigned pa = (unsigned)f2bf_bits(r0[j].x) | ((unsigned)f2bf_bits(r1[j].x) << 16);
          unsigned pb = (unsigned)f2bf_bits(r0[j].y) | ((unsigned)f2bf_bits(r1[j].y) << 16);
          *(unsigned*)&Bs[nc + j * 2    ][kp] = pa;
          *(unsigned*)&Bs[nc + j * 2 + 1][kp] = pb;
        }
      }
    } else {
      // ---- slow path: guarded scalar loads (edge tiles only) ----
      {
        int row = tid >> 1;
        int ks  = (tid & 1) * 32;
        int gm  = m0 + row;
        #pragma unroll 4
        for (int j = 0; j < 32; ++j) {
          int gk = k0 + ks + j;
          float v = (gm < M && gk < K) ? A[(size_t)gm * lda + gk] : 0.f;
          As[row][ks + j] = f2bf_bits(v);
        }
      }
      {
        int k  = tid >> 2;
        int nc = (tid & 3) * 16;
        int gk = k0 + k;
        #pragma unroll 4
        for (int j = 0; j < 16; ++j) {
          int gn = n0 + nc + j;
          float v = (gn < N && gk < K) ? B[(size_t)gk * ldb + gn] : 0.f;
          Bs[nc + j][k] = f2bf_bits(v);
        }
      }
    }
    __syncthreads();

    #pragma unroll
    for (int ks = 0; ks < 2; ++ks) {
      const int ko = ks * 32 + kb;
      bf16x16 a0 = frag_from_lds(&As[wm + mrow     ][ko]);
      bf16x16 a1 = frag_from_lds(&As[wm + 16 + mrow][ko]);
      bf16x16 b0 = frag_from_lds(&Bs[wn + mrow     ][ko]);
      bf16x16 b1 = frag_from_lds(&Bs[wn + 16 + mrow][ko]);
      acc00 = __builtin_amdgcn_wmma_f32_16x16x32_bf16(false, a0, false, b0, (short)0, acc00, false, false);
      acc01 = __builtin_amdgcn_wmma_f32_16x16x32_bf16(false, a0, false, b1, (short)0, acc01, false, false);
      acc10 = __builtin_amdgcn_wmma_f32_16x16x32_bf16(false, a1, false, b0, (short)0, acc10, false, false);
      acc11 = __builtin_amdgcn_wmma_f32_16x16x32_bf16(false, a1, false, b1, (short)0, acc11, false, false);
    }
    __syncthreads();
  }

  // --- epilogue: lanes 0-15 -> M=r, N=lane; lanes 16-31 -> M=8+r, N=lane-16
  const int mb   = m0 + wm + ((lane >> 4) << 3);
  const int ncol = lane & 15;
  const floatx8* accs[4] = { &acc00, &acc01, &acc10, &acc11 };

  if (fullM && fullN) {
    #pragma unroll
    for (int i = 0; i < 2; ++i) {
      #pragma unroll
      for (int j = 0; j < 2; ++j) {
        const floatx8& acc = *accs[i * 2 + j];
        int gn = n0 + wn + j * 16 + ncol;
        float bv = (flags & GF_BIAS) ? bias[gn] : 0.f;
        #pragma unroll
        for (int r = 0; r < 8; ++r) {
          int gm = mb + i * 16 + r;
          float v = acc[r] + bv;
          if (flags & GF_ACC)  v += C[(size_t)gm * ldc + gn];
          if (flags & GF_RELU) v = fmaxf(v, 0.f);
          C[(size_t)gm * ldc + gn] = v;
        }
      }
    }
  } else {
    #pragma unroll
    for (int i = 0; i < 2; ++i) {
      #pragma unroll
      for (int j = 0; j < 2; ++j) {
        const floatx8& acc = *accs[i * 2 + j];
        int gn = n0 + wn + j * 16 + ncol;
        if (gn >= N) continue;
        float bv = (flags & GF_BIAS) ? bias[gn] : 0.f;
        #pragma unroll
        for (int r = 0; r < 8; ++r) {
          int gm = mb + i * 16 + r;
          if (gm >= M) continue;
          float v = acc[r] + bv;
          if (flags & GF_ACC)  v += C[(size_t)gm * ldc + gn];
          if (flags & GF_RELU) v = fmaxf(v, 0.f);
          C[(size_t)gm * ldc + gn] = v;
        }
      }
    }
  }
}

// ---------------------------------------------------------------------------
// One-time weight repack: out[a][k*64+u] = in[k*kstride + (a+aoff)*64 + u]
// Turns (K=16, A, 64) expert weights into (A, 1024) stacked matrices.
__global__ __launch_bounds__(256)
void repack_kernel(float* __restrict__ out, const float* __restrict__ in,
                   int Arows, int kstride, int aoff) {
  int idx = blockIdx.x * 256 + threadIdx.x;
  if (idx >= Arows * 1024) return;
  int a = idx >> 10;
  int j = idx & 1023;
  int k = j >> 6, u = j & 63;
  out[idx] = in[(size_t)k * kstride + (size_t)(a + aoff) * 64 + u];
}

// gfeat[c] = mean over 7x7 of features3[0,c,:,:]
__global__ void meanpool_kernel(float* __restrict__ out, const float* __restrict__ f3) {
  int c = threadIdx.x;               // 256 channels, 256 threads
  float s = 0.f;
  for (int i = 0; i < 49; ++i) s += f3[c * 49 + i];
  out[c] = s * (1.f / 49.f);
}

// out[n] = sum_k in[k*N + n]
__global__ void sumbias_kernel(float* __restrict__ out, const float* __restrict__ in,
                               int Kk, int N) {
  int n = blockIdx.x * 256 + threadIdx.x;
  if (n < N) {
    float s = 0.f;
    for (int k = 0; k < Kk; ++k) s += in[(size_t)k * N + n];
    out[n] = s;
  }
}

__global__ void copy_kernel(float* __restrict__ d, const float* __restrict__ s, int n) {
  int i = blockIdx.x * 256 + threadIdx.x;
  if (i < n) d[i] = s[i];
}

// adj[p] = wts[p,:] . Wadj + badj   (24576 rows of 1024; one wave per row)
__global__ __launch_bounds__(256)
void gemv_adj_kernel(float* __restrict__ adj, const float* __restrict__ wts,
                     const float* __restrict__ Wadj, const float* __restrict__ badj) {
  int row  = blockIdx.x * 8 + (threadIdx.x >> 5);
  int lane = threadIdx.x & 31;
  const float* wp = wts + (size_t)row * 1024;
  float s = 0.f;
  for (int r = lane; r < 1024; r += 32) s += wp[r] * Wadj[r];
  #pragma unroll
  for (int o = 16; o > 0; o >>= 1) s += __shfl_xor(s, o, 32);
  if (lane == 0) adj[row] = s + badj[0];
}

// out[i*cols+j] = softmax_j( in[i*rs + j*cs] )   (block per row)
__global__ __launch_bounds__(256)
void softmax_kernel(float* __restrict__ out, const float* __restrict__ in,
                    int cols, int rs, int cs) {
  __shared__ float red[256];
  int i = blockIdx.x, t = threadIdx.x;
  float mx = -1e30f;
  for (int j = t; j < cols; j += 256) mx = fmaxf(mx, in[(size_t)i * rs + (size_t)j * cs]);
  red[t] = mx; __syncthreads();
  for (int s = 128; s > 0; s >>= 1) { if (t < s) red[t] = fmaxf(red[t], red[t + s]); __syncthreads(); }
  mx = red[0]; __syncthreads();
  float sm = 0.f;
  for (int j = t; j < cols; j += 256) sm += __expf(in[(size_t)i * rs + (size_t)j * cs] - mx);
  red[t] = sm; __syncthreads();
  for (int s = 128; s > 0; s >>= 1) { if (t < s) red[t] += red[t + s]; __syncthreads(); }
  sm = red[0]; __syncthreads();
  float inv = 1.f / sm;
  for (int j = t; j < cols; j += 256)
    out[(size_t)i * cols + j] = __expf(in[(size_t)i * rs + (size_t)j * cs] - mx) * inv;
}

// m[p,j] = relu( (ph[(p/hdiv)*1024+j] + po[(p%384)*1024+j] + b1[j]) * s[p,j] )
// 1024-wide (all 16 experts fused); may run in place (m == s).
__global__ __launch_bounds__(256)
void mulrelu_kernel(float* __restrict__ m, const float* __restrict__ s,
                    const float* __restrict__ ph, int hdiv,
                    const float* __restrict__ po, const float* __restrict__ b1) {
  size_t idx = (size_t)blockIdx.x * 256 + threadIdx.x;
  int p = (int)(idx >> 10), j = (int)(idx & 1023);
  float a = 0.f;
  if (ph) a += ph[(size_t)(p / hdiv) * 1024 + j];
  if (po) a += po[(size_t)(p % 384) * 1024 + j];
  if (b1) a += b1[j];
  m[idx] = fmaxf(a * s[idx], 0.f);
}

// out[i,r] = sum_j w[i*J+j] * Mb[(i*si + j*sj)*1024 + r]
__global__ __launch_bounds__(256)
void wreduce_kernel(float* __restrict__ out, const float* __restrict__ w,
                    const float* __restrict__ Mb, int J, int si, int sj) {
  int i = blockIdx.y;
  int r = blockIdx.x * 256 + threadIdx.x;
  float acc = 0.f;
  for (int j = 0; j < J; ++j)
    acc += w[(size_t)i * J + j] * Mb[((size_t)(i * si + j * sj)) * 1024 + r];
  out[(size_t)i * 1024 + r] = acc;
}

// xo[i,:] = LayerNorm( xi[i,:] + relu(acc[i,:]) ) * g + b       (1024 wide)
__global__ __launch_bounds__(256)
void addreluln_kernel(float* __restrict__ xo, const float* __restrict__ xi,
                      const float* __restrict__ acc,
                      const float* __restrict__ g, const float* __restrict__ b) {
  __shared__ float red[256];
  int i = blockIdx.x, t = threadIdx.x;
  float v[4];
  #pragma unroll
  for (int c = 0; c < 4; ++c) {
    int j = t + c * 256;
    v[c] = xi[(size_t)i * 1024 + j] + fmaxf(acc[(size_t)i * 1024 + j], 0.f);
  }
  float s = v[0] + v[1] + v[2] + v[3];
  red[t] = s; __syncthreads();
  for (int st = 128; st > 0; st >>= 1) { if (t < st) red[t] += red[t + st]; __syncthreads(); }
  float mean = red[0] * (1.f / 1024.f); __syncthreads();
  float q = 0.f;
  #pragma unroll
  for (int c = 0; c < 4; ++c) { float d = v[c] - mean; q += d * d; }
  red[t] = q; __syncthreads();
  for (int st = 128; st > 0; st >>= 1) { if (t < st) red[t] += red[t + st]; __syncthreads(); }
  float inv = rsqrtf(red[0] * (1.f / 1024.f) + 1e-5f);
  #pragma unroll
  for (int c = 0; c < 4; ++c) {
    int j = t + c * 256;
    xo[(size_t)i * 1024 + j] = (v[c] - mean) * inv * g[j] + b[j];
  }
}

// ---------------------------------------------------------------------------
// host-side helpers
// ---------------------------------------------------------------------------
static inline void gemm(hipStream_t s, const float* A, const float* B, float* C,
                        const float* bias, int M, int N, int K,
                        int lda, int ldb, int ldc, int flags) {
  dim3 g((N + 63) / 64, (M + 127) / 128);
  gemm_wmma_kernel<<<g, 256, 0, s>>>(A, B, C, bias, M, N, K, lda, ldb, ldc, flags);
}

static inline void mulrelu(hipStream_t s, float* m, const float* sv,
                           const float* ph, int hdiv, const float* po, const float* b1) {
  mulrelu_kernel<<<(24576 * 1024) / 256, 256, 0, s>>>(m, sv, ph, hdiv, po, b1);
}

static inline void repack(hipStream_t s, float* out, const float* in,
                          int Arows, int kstride, int aoff) {
  repack_kernel<<<(Arows * 1024 + 255) / 256, 256, 0, s>>>(out, in, Arows, kstride, aoff);
}

// out[24576, ldc] = relu( relu((h@W1t' + o@W1b' + b1) * (sp@W2' + b2)) @ W3stk + b3 )
static void mbf_pair(hipStream_t s, const float* h, const float* o, float* out, int ldc,
                     const float* w1t, const float* w1b, const float* w2,
                     const float* b1flat, const float* b2flat,
                     const float* w3stk, const float* b3sum, const float* sp,
                     float* haAll, float* oaAll, float* SM) {
  gemm(s, h,  w1t, haAll, nullptr, 64,    1024, 1024, 1024, 1024, 1024, 0);
  gemm(s, o,  w1b, oaAll, nullptr, 384,   1024, 1024, 1024, 1024, 1024, 0);
  gemm(s, sp, w2,  SM,    b2flat,  24576, 1024, 1024, 1024, 1024, 1024, GF_BIAS);
  mulrelu(s, SM, SM, haAll, 384, oaAll, b1flat);
  gemm(s, SM, w3stk, out, b3sum,   24576, 1024, 1024, 1024, 1024, ldc, GF_BIAS | GF_RELU);
}

// ---------------------------------------------------------------------------
extern "C" void kernel_launch(void* const* d_in, const int* in_sizes, int n_in,
                              void* d_out, int out_size, void* d_ws, size_t ws_size,
                              hipStream_t stream) {
  (void)in_sizes; (void)n_in; (void)out_size; (void)ws_size;

  const float* box  = (const float*)d_in[0];
  const float* sraw = (const float*)d_in[1];
  const float* f3   = (const float*)d_in[2];
  const float* Wb1  = (const float*)d_in[3];
  const float* bb1  = (const float*)d_in[4];
  const float* Wb2  = (const float*)d_in[5];
  const float* bb2  = (const float*)d_in[6];
  const float* Ws1  = (const float*)d_in[7];
  const float* bs1  = (const float*)d_in[8];
  const float* Ws2  = (const float*)d_in[9];
  const float* bs2  = (const float*)d_in[10];
  const float* Ws3  = (const float*)d_in[11];
  const float* bs3  = (const float*)d_in[12];
  const float* Wadj = (const float*)d_in[13];
  const float* badj = (const float*)d_in[14];
  const float* gh   = (const float*)d_in[15];
  const float* bh   = (const float*)d_in[16];
  const float* go   = (const float*)d_in[17];
  const float* bo   = (const float*)d_in[18];
  const float* A1   = (const float*)d_in[19];
  const float* Ab1  = (const float*)d_in[20];   // (16,64) == flat 1024 bias
  const float* A2   = (const float*)d_in[21];
  const float* Ab2  = (const float*)d_in[22];
  const float* A3   = (const float*)d_in[23];   // (16,64,1024) == stacked (1024,1024)
  const float* Ab3  = (const float*)d_in[24];
  const float* G1   = (const float*)d_in[25];
  const float* Gb1  = (const float*)d_in[26];
  const float* G2   = (const float*)d_in[27];
  const float* Gb2  = (const float*)d_in[28];
  const float* G3   = (const float*)d_in[29];
  const float* Gb3  = (const float*)d_in[30];
  const float* S1   = (const float*)d_in[31];
  const float* Sb1  = (const float*)d_in[32];
  const float* S2   = (const float*)d_in[33];
  const float* Sb2  = (const float*)d_in[34];
  const float* S3   = (const float*)d_in[35];
  const float* Sb3  = (const float*)d_in[36];
  const float* O1   = (const float*)d_in[37];
  const float* Ob1  = (const float*)d_in[38];
  const float* O2   = (const float*)d_in[39];
  const float* Ob2  = (const float*)d_in[40];
  const float* O3   = (const float*)d_in[41];
  const float* Ob3  = (const float*)d_in[42];
  // d_in[43] = n_h (==64, hard-coded to match compile-time tiling)

  const int P = 24576;   // NH*NB

  float* w = (float*)d_ws;
  size_t off = 0;
  auto alloc = [&](size_t n) { float* p = w + off; off += n; return p; };
  float* sp    = alloc((size_t)P * 1024);   // spatial features
  float* Mbuf  = alloc((size_t)P * 1024);   // wts / m_o2s / m_s2o
  float* SM    = alloc((size_t)P * 1024);   // s_all -> m_all (in place)
  float* wA1t  = alloc(1024 * 1024);        // repacked expert weights
  float* wA1b  = alloc(1024 * 1024);
  float* wA2   = alloc(1024 * 1024);
  float* wO1   = alloc(1024 * 1024);
  float* wO2   = alloc(1024 * 1024);
  float* wS1   = alloc(1024 * 1024);
  float* wS2   = alloc(1024 * 1024);
  float* wG1   = alloc(256 * 1024);
  float* wG2   = alloc(1024 * 1024);
  float* haAll = alloc(64 * 1024);          // ha/ah/ga (all experts)
  float* oaAll = alloc(384 * 1024);         // oa/ao
  float* hA    = alloc(64 * 1024);
  float* hB    = alloc(64 * 1024);
  float* oA    = alloc(384 * 1024);
  float* oB    = alloc(384 * 1024);
  float* adj   = alloc(64 * 384);
  float* w1s   = alloc(64 * 384);
  float* w2s   = alloc(384 * 64);
  float* hacc  = alloc(64 * 1024);
  float* oacc  = alloc(384 * 1024);
  float* gfeat = alloc(256);
  float* b3A   = alloc(1024);
  float* b3G   = alloc(1024);
  float* b3S   = alloc(1024);
  float* b3O   = alloc(1024);
  // transient overlays inside Mbuf (dead before Mbuf's first real use):
  float* enc1 = Mbuf;                       // 384x1024
  float* sp1  = Mbuf;                       // 24576x128
  float* sp2  = Mbuf + (size_t)P * 128;     // 24576x256

  float* out  = (float*)d_out;              // (24576, 2048)

  // --- one-time precomputes ---
  meanpool_kernel<<<1, 256, 0, stream>>>(gfeat, f3);
  sumbias_kernel<<<4, 256, 0, stream>>>(b3A, Ab3, 16, 1024);
  sumbias_kernel<<<4, 256, 0, stream>>>(b3G, Gb3, 16, 1024);
  sumbias_kernel<<<4, 256, 0, stream>>>(b3S, Sb3, 16, 1024);
  sumbias_kernel<<<4, 256, 0, stream>>>(b3O, Ob3, 16, 1024);
  repack(stream, wA1t, A1, 1024, 2048 * 64, 0);
  repack(stream, wA1b, A1, 1024, 2048 * 64, 1024);
  repack(stream, wA2,  A2, 1024, 1024 * 64, 0);
  repack(stream, wO1,  O1, 1024, 1024 * 64, 0);
  repack(stream, wO2,  O2, 1024, 1024 * 64, 0);
  repack(stream, wS1,  S1, 1024, 1024 * 64, 0);
  repack(stream, wS2,  S2, 1024, 1024 * 64, 0);
  repack(stream, wG1,  G1, 256,  256 * 64,  0);
  repack(stream, wG2,  G2, 1024, 1024 * 64, 0);

  // --- box MLP: enc = relu(relu(X@Wb1+b)@Wb2+b) ---
  gemm(stream, box,  Wb1, enc1, bb1, 384, 1024, 12544, 12544, 1024, 1024, GF_BIAS | GF_RELU);
  gemm(stream, enc1, Wb2, oA,   bb2, 384, 1024, 1024,  1024,  1024, 1024, GF_BIAS | GF_RELU);
  copy_kernel<<<(64 * 1024) / 256, 256, 0, stream>>>(hA, oA, 64 * 1024);

  // --- spatial MLP: 36 -> 128 -> 256 -> 1024 ---
  gemm(stream, sraw, Ws1, sp1, bs1, P, 128,  36,  36,  128,  128,  GF_BIAS | GF_RELU);
  gemm(stream, sp1,  Ws2, sp2, bs2, P, 256,  128, 128, 256,  256,  GF_BIAS | GF_RELU);
  gemm(stream, sp2,  Ws3, sp,  bs3, P, 1024, 256, 256, 1024, 1024, GF_BIAS | GF_RELU);

  float* hc = hA; float* hn = hB; float* oc = oA; float* on = oB;

  for (int it = 0; it < 2; ++it) {
    // wts -> Mbuf
    mbf_pair(stream, hc, oc, Mbuf, 1024, wA1t, wA1b, wA2, Ab1, Ab2, A3, b3A, sp,
             haAll, oaAll, SM);
    gemv_adj_kernel<<<P / 8, 256, 0, stream>>>(adj, Mbuf, Wadj, badj);
    softmax_kernel<<<64, 256, 0, stream>>>(w1s, adj, 384, 384, 1);

    // m_o2s -> Mbuf   (ao = o@O1'+Ob1 ; so = sp@O2'+Ob2 ; relu(ao*so)@O3 + b3O)
    gemm(stream, oc, wO1, oaAll, Ob1, 384, 1024, 1024, 1024, 1024, 1024, GF_BIAS);
    gemm(stream, sp, wO2, SM,    Ob2, P,   1024, 1024, 1024, 1024, 1024, GF_BIAS);
    mulrelu(stream, SM, SM, nullptr, 1, oaAll, nullptr);
    gemm(stream, SM, O3, Mbuf, b3O, P, 1024, 1024, 1024, 1024, 1024, GF_BIAS);
    wreduce_kernel<<<dim3(4, 64), 256, 0, stream>>>(hacc, w1s, Mbuf, 384, 384, 1);
    addreluln_kernel<<<64, 256, 0, stream>>>(hn, hc, hacc, gh, bh);

    softmax_kernel<<<384, 256, 0, stream>>>(w2s, adj, 64, 1, 384);

    // m_s2o -> Mbuf   (uses updated h)
    gemm(stream, hn, wS1, haAll, Sb1, 64, 1024, 1024, 1024, 1024, 1024, GF_BIAS);
    gemm(stream, sp, wS2, SM,    Sb2, P,  1024, 1024, 1024, 1024, 1024, GF_BIAS);
    mulrelu(stream, SM, SM, haAll, 384, nullptr, nullptr);
    gemm(stream, SM, S3, Mbuf, b3S, P, 1024, 1024, 1024, 1024, 1024, GF_BIAS);
    wreduce_kernel<<<dim3(4, 384), 256, 0, stream>>>(oacc, w2s, Mbuf, 64, 1, 384);
    addreluln_kernel<<<384, 256, 0, stream>>>(on, oc, oacc, go, bo);

    { float* t = hc; hc = hn; hn = t; }
    { float* t = oc; oc = on; on = t; }
  }

  // --- attn -> d_out[:, 0:1024] ---
  mbf_pair(stream, hc, oc, out, 2048, wA1t, wA1b, wA2, Ab1, Ab2, A3, b3A, sp,
           haAll, oaAll, SM);

  // --- attn_g -> d_out[:, 1024:2048]  (app row = gfeat broadcast) ---
  gemm(stream, gfeat, wG1, haAll, nullptr, 1, 1024, 256, 256, 1024, 1024, 0);
  gemm(stream, sp,    wG2, SM,    Gb2,     P, 1024, 1024, 1024, 1024, 1024, GF_BIAS);
  mulrelu(stream, SM, SM, haAll, 1 << 30, nullptr, Gb1);
  gemm(stream, SM, G3, out + 1024, b3G, P, 1024, 1024, 1024, 1024, 2048, GF_BIAS | GF_RELU);
}
